// NonLocalDenoise_6700148982065
// MI455X (gfx1250) — matrix operand order repaired and checked by reading, not compiled, not run on hardware
//
#include <hip/hip_runtime.h>
#include <hip/hip_bf16.h>
#include <math.h>

typedef __attribute__((ext_vector_type(16))) _Float16 v16h;
typedef __attribute__((ext_vector_type(8)))  float    v8f;

#define SDIM 4096
#define CCH  256
#define DQK  128
#define BR   128
#define BC   32
#define NBATCH 8

__device__ __forceinline__ v8f wmma_f16(v16h a, v16h b, v8f c) {
  return __builtin_amdgcn_wmma_f32_16x16x32_f16(false, a, false, b, (short)0, c,
                                                false, false);
}

union HFrag { v16h v; uint4 u[2]; };

// A fragment (16x32 f16, rows m x cols k), row stride ld halves.
// ISA 7.12.2: lanes 0-15: m=lane, K={kbase+0..7, kbase+16..23};
// lanes 16-31: m=lane-16, K={kbase+8..15, kbase+24..31}.
__device__ __forceinline__ v16h load_a_frag(const _Float16* base, int ld,
                                            int kbase, int lane) {
  int m = lane & 15, h = lane >> 4;
  const _Float16* p0 = base + (size_t)m * ld + kbase + h * 8;
  HFrag f;
  f.u[0] = *(const uint4*)p0;
  f.u[1] = *(const uint4*)(p0 + 16);
  return f.v;
}

// B fragment (32x16 f16) where B[k][n] = M[nbase+n][kbase+k], row stride ld.
// Lanes 0-15 (n=lane) hold K={0..7,16..23}, lanes 16-31 hold K={8..15,24..31}.
__device__ __forceinline__ v16h load_b_frag(const _Float16* M, int ld,
                                            int nbase, int kbase, int lane) {
  int n = lane & 15, h = lane >> 4;
  const _Float16* p = M + (size_t)(nbase + n) * ld + kbase + h * 8;
  HFrag f;
  f.u[0] = *(const uint4*)p;
  f.u[1] = *(const uint4*)(p + 16);
  return f.v;
}

// ---- CDNA5 async global->LDS copy (ASYNCcnt-tracked, no VGPR round-trip) ----
__device__ __forceinline__ uint32_t lds_addr(const void* p) {
  // Generic LDS addresses carry the LDS byte offset in the low 32 bits.
  return (uint32_t)(uintptr_t)p;
}
__device__ __forceinline__ void async_b128(uint32_t lds_byte, const void* g) {
  asm volatile("global_load_async_to_lds_b128 %0, %1, off"
               :: "v"(lds_byte), "v"(g) : "memory");
}
template <int N>
__device__ __forceinline__ void wait_async() {
  asm volatile("s_wait_asynccnt %0" :: "n"(N) : "memory");
}

// ---- DPP16 butterfly reductions over a 16-lane row (no LDS traffic) ----
template <int CTRL>
__device__ __forceinline__ float f_dpp(float v) {
  int r = __builtin_amdgcn_update_dpp(0, __builtin_bit_cast(int, v), CTRL, 0xf,
                                      0xf, true);
  return __builtin_bit_cast(float, r);
}
__device__ __forceinline__ float red_max16(float v) {
  v = fmaxf(v, f_dpp<0xB1>(v));   // quad_perm(1,0,3,2)  : xor1
  v = fmaxf(v, f_dpp<0x4E>(v));   // quad_perm(2,3,0,1)  : xor2
  v = fmaxf(v, f_dpp<0x141>(v));  // row_half_mirror     : xor4 (post-reduce)
  v = fmaxf(v, f_dpp<0x140>(v));  // row_mirror          : xor8 (post-reduce)
  return v;
}
__device__ __forceinline__ float red_sum16(float v) {
  v += f_dpp<0xB1>(v);
  v += f_dpp<0x4E>(v);
  v += f_dpp<0x141>(v);
  v += f_dpp<0x140>(v);
  return v;
}

// ---------------------------------------------------------------------------
// Projection: Q[n][s][o] = sum_c theta_w[o][c]*x[n][c][s] (f16 out), same for K.
// ---------------------------------------------------------------------------
__global__ void __launch_bounds__(256)
proj_kernel(const float* __restrict__ x, const float* __restrict__ wt,
            const float* __restrict__ wp, _Float16* __restrict__ Qg,
            _Float16* __restrict__ Kg) {
  __shared__ float xs[32][33];
  __shared__ float wts[128][33];
  __shared__ float wps[128][33];
  const int n = blockIdx.y;
  const int s0 = blockIdx.x * 32;
  const int tid = threadIdx.x;
  const int o = tid & 127;
  const int half = tid >> 7;

  float accq[16], acck[16];
#pragma unroll
  for (int j = 0; j < 16; ++j) { accq[j] = 0.0f; acck[j] = 0.0f; }

  for (int c0 = 0; c0 < CCH; c0 += 32) {
    {
      int r = tid >> 3, cc = (tid & 7) * 4;
      float4 v = *(const float4*)(x + (((size_t)n * CCH + c0 + r) * SDIM + s0 + cc));
      xs[r][cc + 0] = v.x; xs[r][cc + 1] = v.y;
      xs[r][cc + 2] = v.z; xs[r][cc + 3] = v.w;
    }
    {
      int r = tid >> 1, off = (tid & 1) * 16;
      const float4* st_ = (const float4*)(wt + ((size_t)r * CCH + c0 + off));
      const float4* sp_ = (const float4*)(wp + ((size_t)r * CCH + c0 + off));
#pragma unroll
      for (int i = 0; i < 2; ++i) {
        float4 a = st_[i], b = sp_[i];
        wts[r][off + i * 4 + 0] = a.x; wts[r][off + i * 4 + 1] = a.y;
        wts[r][off + i * 4 + 2] = a.z; wts[r][off + i * 4 + 3] = a.w;
        wps[r][off + i * 4 + 0] = b.x; wps[r][off + i * 4 + 1] = b.y;
        wps[r][off + i * 4 + 2] = b.z; wps[r][off + i * 4 + 3] = b.w;
      }
    }
    __syncthreads();
    for (int cc = 0; cc < 32; ++cc) {
      float wq = wts[o][cc], wk = wps[o][cc];
#pragma unroll
      for (int j = 0; j < 16; ++j) {
        float xv = xs[cc][half + 2 * j];
        accq[j] = fmaf(wq, xv, accq[j]);
        acck[j] = fmaf(wk, xv, acck[j]);
      }
    }
    __syncthreads();
  }
#pragma unroll
  for (int j = 0; j < 16; ++j) {
    int s = s0 + half + 2 * j;
    size_t idx = ((size_t)n * SDIM + s) * DQK + o;
    Qg[idx] = (_Float16)accq[j];
    Kg[idx] = (_Float16)acck[j];
  }
}

// ---------------------------------------------------------------------------
// f16 copy of x -> Vt (layout [n][C][S] == x's layout, i.e. V transposed)
// ---------------------------------------------------------------------------
__global__ void __launch_bounds__(256)
cvt_kernel(const float* __restrict__ x, _Float16* __restrict__ vt) {
  size_t i = ((size_t)blockIdx.x * blockDim.x + threadIdx.x) * 4;
  float4 v = *(const float4*)(x + i);
  union { _Float16 h[4]; uint2 u; } r;
  r.h[0] = (_Float16)v.x; r.h[1] = (_Float16)v.y;
  r.h[2] = (_Float16)v.z; r.h[3] = (_Float16)v.w;
  *(uint2*)(vt + i) = r.u;
}

// ---------------------------------------------------------------------------
// Fused flash attention: out = x + softmax(Q K^T) V (residual included).
// 256 threads = 8 waves; wave w owns Q rows [blk*128 + w*16, +16).
// Double-buffered async K/V tiles; 24 v_wmma_f32_16x16x32_f16 per iteration.
// ---------------------------------------------------------------------------
__device__ __forceinline__ void prefetch_kv(const _Float16* Kg,
                                            const _Float16* Vg, int kv0,
                                            _Float16* Kl, _Float16* Vl,
                                            int tid) {
  // K tile: contiguous 8 KB; 2 x b128 per thread.
  const _Float16* ks = Kg + (size_t)kv0 * DQK + tid * 8;
  async_b128(lds_addr(Kl + tid * 8), ks);
  async_b128(lds_addr(Kl + tid * 8 + 2048), ks + 2048);
  // Vt tile: 256 rows x 64 B; 4 x b128 per thread (thread == channel row).
  const _Float16* vs = Vg + (size_t)tid * SDIM + kv0;
#pragma unroll
  for (int i = 0; i < 4; ++i)
    async_b128(lds_addr(Vl + tid * BC + i * 8), vs + i * 8);
}

__global__ void __launch_bounds__(256)
attn_kernel(const float* __restrict__ x, const _Float16* __restrict__ Q,
            const _Float16* __restrict__ K, const _Float16* __restrict__ Vt,
            float* __restrict__ out) {
  __shared__ __align__(16) _Float16 Klds[2][BC * DQK];   // 2 x 8 KB
  __shared__ __align__(16) _Float16 Vlds[2][CCH * BC];   // 2 x 16 KB
  __shared__ __align__(16) _Float16 Plds[8 * 16 * BC];   // 8 KB

  const int tid = threadIdx.x;
  const int wave = tid >> 5;
  const int lane = tid & 31;
  const int lh = lane >> 4;
  const int ln = lane & 15;
  const int n = blockIdx.y;
  const int qrow = blockIdx.x * BR + wave * 16;

  const _Float16* Qg = Q + ((size_t)n * SDIM + qrow) * DQK;
  const _Float16* Kgp = K + (size_t)n * SDIM * DQK;
  const _Float16* Vg = Vt + (size_t)n * CCH * SDIM;

  // Q strip resident in registers for the whole pass (4 A fragments, d=128)
  v16h qf[4];
#pragma unroll
  for (int kc = 0; kc < 4; ++kc) qf[kc] = load_a_frag(Qg, DQK, kc * 32, lane);

  v8f o[16];
#pragma unroll
  for (int ct = 0; ct < 16; ++ct)
#pragma unroll
    for (int r = 0; r < 8; ++r) o[ct][r] = 0.0f;

  float mreg[8], lreg[8];
#pragma unroll
  for (int r = 0; r < 8; ++r) { mreg[r] = -INFINITY; lreg[r] = 0.0f; }

  _Float16* Pw = Plds + wave * (16 * BC);
  const int NT = SDIM / BC;

  prefetch_kv(Kgp, Vg, 0, Klds[0], Vlds[0], tid);

  for (int it = 0; it < NT; ++it) {
    const int cur = it & 1;
    const _Float16* Kl = Klds[cur];
    const _Float16* Vl = Vlds[cur];

    if (it + 1 < NT) {
      prefetch_kv(Kgp, Vg, (it + 1) * BC, Klds[cur ^ 1], Vlds[cur ^ 1], tid);
      wait_async<6>();   // oldest 6 (current tile) retired; next 6 in flight
    } else {
      wait_async<0>();
    }
    __syncthreads();

    // S = Q K^T : batch all 8 B fragments, then 8 back-to-back WMMAs
    v16h kb[8];
#pragma unroll
    for (int kc = 0; kc < 4; ++kc) {
      kb[2 * kc + 0] = load_b_frag(Kl, DQK, 0, kc * 32, lane);
      kb[2 * kc + 1] = load_b_frag(Kl, DQK, 16, kc * 32, lane);
    }
    v8f st[2];
#pragma unroll
    for (int t = 0; t < 2; ++t)
#pragma unroll
      for (int r = 0; r < 8; ++r) st[t][r] = 0.0f;
#pragma unroll
    for (int kc = 0; kc < 4; ++kc) {
      st[0] = wmma_f16(qf[kc], kb[2 * kc + 0], st[0]);
      st[1] = wmma_f16(qf[kc], kb[2 * kc + 1], st[1]);
    }

    // online softmax: C-layout row = r + lh*8; DPP16 reductions stay inside
    // the 16-lane row that owns the matrix row.
#pragma unroll
    for (int r = 0; r < 8; ++r) {
      float v = red_max16(fmaxf(st[0][r], st[1][r]));
      float mn = fmaxf(mreg[r], v);
      float sc = __expf(mreg[r] - mn);
      mreg[r] = mn;
      st[0][r] = __expf(st[0][r] - mn);
      st[1][r] = __expf(st[1][r] - mn);
      float sum = red_sum16(st[0][r] + st[1][r]);
      lreg[r] = lreg[r] * sc + sum;
#pragma unroll
      for (int ct = 0; ct < 16; ++ct) o[ct][r] *= sc;
    }

    // stage P (C-layout fp32 -> f16 row-major in LDS), reload as A fragment
#pragma unroll
    for (int r = 0; r < 8; ++r) {
      int row = r + lh * 8;
      Pw[row * BC + ln] = (_Float16)st[0][r];
      Pw[row * BC + 16 + ln] = (_Float16)st[1][r];
    }
    v16h pa = load_a_frag(Pw, BC, 0, lane);

    // O += P V : 16 independent WMMAs, B fragments grouped 4-at-a-time
#pragma unroll
    for (int g = 0; g < 4; ++g) {
      v16h vb[4];
#pragma unroll
      for (int j = 0; j < 4; ++j)
        vb[j] = load_b_frag(Vl, BC, (g * 4 + j) * 16, 0, lane);
#pragma unroll
      for (int j = 0; j < 4; ++j)
        o[g * 4 + j] = wmma_f16(pa, vb[j], o[g * 4 + j]);
    }
    __syncthreads();
  }

  // epilogue: transpose each 16x16 O tile through LDS so the residual read
  // and final store are contiguous float4 accesses along s.
  float invl[8];
#pragma unroll
  for (int r = 0; r < 8; ++r) invl[r] = 1.0f / lreg[r];

  float* Tw = ((float*)&Vlds[0][0]) + wave * (16 * 17);  // V buffers now free
  const float* xg = x + (size_t)n * CCH * SDIM;
  float* og = out + (size_t)n * CCH * SDIM;
  const int crow = lane >> 1;        // 0..15 : channel row within tile
  const int soff = (lane & 1) * 8;   // 0 / 8 : s offset within strip

#pragma unroll
  for (int ct = 0; ct < 16; ++ct) {
#pragma unroll
    for (int r = 0; r < 8; ++r)
      Tw[ln * 17 + (r + lh * 8)] = o[ct][r] * invl[r];
    // same-wave LDS ordering makes the stores visible to the loads below
    int c = ct * 16 + crow;
    const float* xp = xg + (size_t)c * SDIM + qrow + soff;
    float* op = og + (size_t)c * SDIM + qrow + soff;
    float4 x0 = *(const float4*)xp;
    float4 x1 = *(const float4*)(xp + 4);
    const float* trow = Tw + crow * 17 + soff;
    float4 y0, y1;
    y0.x = x0.x + trow[0]; y0.y = x0.y + trow[1];
    y0.z = x0.z + trow[2]; y0.w = x0.w + trow[3];
    y1.x = x1.x + trow[4]; y1.y = x1.y + trow[5];
    y1.z = x1.z + trow[6]; y1.w = x1.w + trow[7];
    *(float4*)op = y0;
    *(float4*)(op + 4) = y1;
  }
}

extern "C" void kernel_launch(void* const* d_in, const int* in_sizes, int n_in,
                              void* d_out, int out_size, void* d_ws,
                              size_t ws_size, hipStream_t stream) {
  const float* x = (const float*)d_in[0];
  const float* wt = (const float*)d_in[1];
  const float* wp = (const float*)d_in[2];
  float* out = (float*)d_out;

  char* ws = (char*)d_ws;
  const size_t qkBytes = (size_t)NBATCH * SDIM * DQK * sizeof(_Float16);  // 8 MB
  _Float16* Qg = (_Float16*)(ws);
  _Float16* Kg = (_Float16*)(ws + qkBytes);
  _Float16* Vg = (_Float16*)(ws + 2 * qkBytes);  // 16 MB (f16 copy of x)

  proj_kernel<<<dim3(SDIM / 32, NBATCH), dim3(256), 0, stream>>>(x, wt, wp, Qg, Kg);
  cvt_kernel<<<dim3((NBATCH * CCH * SDIM) / (256 * 4)), dim3(256), 0, stream>>>(x, Vg);
  attn_kernel<<<dim3(SDIM / BR, NBATCH), dim3(256), 0, stream>>>(x, Qg, Kg, Vg, out);
}
